// GONN_59425167507915
// MI455X (gfx1250) — matrix operand order repaired and checked by reading, not compiled, not run on hardware
//
#include <hip/hip_runtime.h>
#include <cstdint>

typedef unsigned short u16;
typedef __attribute__((ext_vector_type(16))) __bf16 bf16x16;
typedef __attribute__((ext_vector_type(8)))  float  f32x8;

struct alignas(16) U128 { unsigned int w[4]; };
union FragBF { bf16x16 v; U128 q[2]; };
struct alignas(16) F4 { float v[4]; };
union Pack8 { U128 q; u16 h[8]; };

#define HDIM   256
#define CDIM   64
#define INDIM  128
#define OUTDIM 64
#define NGRAPH 64

__device__ __forceinline__ u16 f2bf(float f) {
  unsigned int u = __float_as_uint(f);
  unsigned int r = u + 0x7fffu + ((u >> 16) & 1u);
  return (u16)(r >> 16);
}

__device__ __forceinline__ float wsum(float v) {
#pragma unroll
  for (int o = 16; o > 0; o >>= 1) v += __shfl_xor(v, o, 32);
  return v;
}
__device__ __forceinline__ float wmax(float v) {
#pragma unroll
  for (int o = 16; o > 0; o >>= 1) v = fmaxf(v, __shfl_xor(v, o, 32));
  return v;
}

// ---------------- utility kernels ----------------
__global__ void zero_f(float* p, size_t n) {
  size_t i = (size_t)blockIdx.x * blockDim.x + threadIdx.x;
  if (i < n) p[i] = 0.0f;
}

__global__ void cvt_bf16(const float* __restrict__ src, u16* __restrict__ dst, size_t n) {
  size_t i = (size_t)blockIdx.x * blockDim.x + threadIdx.x;
  if (i < n) dst[i] = f2bf(src[i]);
}

// Pack row-major f32 B [K,Ncols] into per-lane WMMA bf16 fragment layout.
// Tile (kt,nt): lane l holds col = nt*16+(l&15), K rows kt*32 + (l&16?16:0) + j, j=0..15.
__global__ void pack_b(const float* __restrict__ B, u16* __restrict__ dst, int K, int Ncols) {
  int tid = blockIdx.x * blockDim.x + threadIdx.x;
  int KT = K >> 5, NT = Ncols >> 4;
  if (tid >= KT * NT * 32) return;
  int lane = tid & 31;
  int tile = tid >> 5;
  int nt = tile % NT, kt = tile / NT;
  int col = nt * 16 + (lane & 15);
  int kst = kt * 32 + ((lane & 16) ? 16 : 0);
  u16* d = dst + (size_t)tid * 16;
#pragma unroll
  for (int j = 0; j < 16; ++j) d[j] = f2bf(B[(size_t)(kst + j) * Ncols + col]);
}

// ---------------- graph aggregation ----------------
__global__ void deg_kernel(const int* __restrict__ ei, float* __restrict__ deg, int E) {
  int e = blockIdx.x * blockDim.x + threadIdx.x;
  if (e >= E) return;
  int s = ei[e], d = ei[E + e];
  if (s != d) atomicAdd(deg + d, 1.0f);
}

__global__ void agg_scatter(const int* __restrict__ ei, const float* __restrict__ xc,
                            float* __restrict__ m, int E) {
  int tid = blockIdx.x * blockDim.x + threadIdx.x;
  int e = tid >> 5, lane = tid & 31;
  if (e >= E) return;
  int s = ei[e], d = ei[E + e];
  if (s == d) return;
  const float* xs = xc + (size_t)s * HDIM + lane * 8;
  float* md = m + (size_t)d * HDIM + lane * 8;
#pragma unroll
  for (int j = 0; j < 8; ++j) atomicAdd(md + j, xs[j]);
}

__global__ void agg_finalize(float* __restrict__ m, u16* __restrict__ mb,
                             const float* __restrict__ xc, const float* __restrict__ deg,
                             size_t total) {
  size_t i = (size_t)blockIdx.x * blockDim.x + threadIdx.x;
  if (i >= total) return;
  float v = (m[i] + xc[i]) / (deg[i >> 8] + 1.0f);
  m[i] = v;
  mb[i] = f2bf(v);
}

// ---------------- WMMA GEMM: out[M,Ncols] = A[M,K](bf16) @ Bpacked + bias ----------------
// A split: k < splitK from A0 (stride lda0), else A1 (stride lda1). One wave per 16x16 tile.
__global__ void gemm_bf16_wmma(const u16* __restrict__ A0, const u16* __restrict__ A1,
                               int splitK, int lda0, int lda1,
                               const u16* __restrict__ Bp, const float* __restrict__ bias,
                               float* __restrict__ out, int ldo, int M, int Ncols, int K) {
  int tid = blockIdx.x * blockDim.x + threadIdx.x;
  int wave = tid >> 5;
  int lane = tid & 31;
  int NT = Ncols >> 4;
  int rt = wave / NT;
  int nt = wave - rt * NT;
  if (rt * 16 >= M) return;
  int row = rt * 16 + (lane & 15);
  if (row >= M) row = M - 1;
  int kbase = (lane & 16) ? 16 : 0;
  f32x8 acc = {};
  int KT = K >> 5;
  for (int kt = 0; kt < KT; ++kt) {
    int k0 = kt << 5;
    const u16* Ap; int kk, ld;
    if (k0 < splitK) { Ap = A0; kk = k0; ld = lda0; }
    else             { Ap = A1; kk = k0 - splitK; ld = lda1; }
    size_t abase = (size_t)row * ld + kk + kbase;
    FragBF fa, fb;
    fa.q[0] = *(const U128*)(Ap + abase);        // K kbase..kbase+7
    fa.q[1] = *(const U128*)(Ap + abase + 8);    // K kbase+8..kbase+15
    const u16* bp = Bp + (((size_t)kt * NT + nt) * 32 + lane) * 16;
    fb.q[0] = *(const U128*)(bp);
    fb.q[1] = *(const U128*)(bp + 8);
    __builtin_prefetch(Ap + abase + 32, 0, 1);   // global_prefetch_b8
    acc = __builtin_amdgcn_wmma_f32_16x16x32_bf16(false, fa.v, false, fb.v,
                                                  (short)0, acc, false, false);
  }
  int col = nt * 16 + (lane & 15);
  float b = bias ? bias[col] : 0.0f;
  int rbase = rt * 16 + ((lane & 16) ? 8 : 0);
#pragma unroll
  for (int r = 0; r < 8; ++r) {
    int orow = rbase + r;
    if (orow < M) out[(size_t)orow * ldo + col] = acc[r] + b;
  }
}

// ---------------- epilogues (one wave32 per node row) ----------------
__global__ void ln_input(const float* __restrict__ h, const float* __restrict__ s,
                         const float* __restrict__ b, float* __restrict__ xc,
                         u16* __restrict__ xcb, int N) {
  int tid = blockIdx.x * blockDim.x + threadIdx.x;
  int row = tid >> 5, lane = tid & 31;
  if (row >= N) return;
  size_t base = (size_t)row * HDIM + lane * 8;
  F4 a = ((const F4*)(h + base))[0], c = ((const F4*)(h + base))[1];
  float y[8], sm = 0.f, sq = 0.f;
#pragma unroll
  for (int j = 0; j < 8; ++j) {
    float v = fmaxf((j < 4) ? a.v[j] : c.v[j - 4], 0.0f);   // relu
    y[j] = v; sm += v; sq += v * v;
  }
  sm = wsum(sm); sq = wsum(sq);
  float mean = sm * (1.0f / HDIM);
  float rstd = rsqrtf(sq * (1.0f / HDIM) - mean * mean + 1e-5f);
  Pack8 pk; F4 o0, o1;
#pragma unroll
  for (int j = 0; j < 8; ++j) {
    int col = lane * 8 + j;
    float v = (y[j] - mean) * rstd * s[col] + b[col];
    if (j < 4) o0.v[j] = v; else o1.v[j - 4] = v;
    pk.h[j] = f2bf(v);
  }
  ((F4*)(xc + base))[0] = o0; ((F4*)(xc + base))[1] = o1;
  *(U128*)(xcb + base) = pk.q;
}

// softmax(64) -> cumsum -> diff_or(tm) -> repeat(4) blend -> layernorm
__global__ void layer_post(const float* __restrict__ logits, float* __restrict__ tm,
                           float* __restrict__ xc, u16* __restrict__ xcb,
                           const float* __restrict__ m, const float* __restrict__ lns,
                           const float* __restrict__ lnb, int N) {
  int tid = blockIdx.x * blockDim.x + threadIdx.x;
  int row = tid >> 5, lane = tid & 31;
  if (row >= N) return;
  float l0 = logits[(size_t)row * CDIM + 2 * lane];
  float l1 = logits[(size_t)row * CDIM + 2 * lane + 1];
  float mx = wmax(fmaxf(l0, l1));
  float e0 = __expf(l0 - mx), e1 = __expf(l1 - mx);
  float s = wsum(e0 + e1);
  float p0 = e0 / s, p1 = e1 / s;
  float pair = p0 + p1, inc = pair;
#pragma unroll
  for (int o = 1; o < 32; o <<= 1) { float t = __shfl_up(inc, o, 32); if (lane >= o) inc += t; }
  float excl = inc - pair;
  float c0 = excl + p0, c1 = excl + p0 + p1;
  float t0 = tm[(size_t)row * CDIM + 2 * lane];
  float t1 = tm[(size_t)row * CDIM + 2 * lane + 1];
  float r0 = t0 + (1.0f - t0) * c0;
  float r1 = t1 + (1.0f - t1) * c1;
  tm[(size_t)row * CDIM + 2 * lane]     = r0;
  tm[(size_t)row * CDIM + 2 * lane + 1] = r1;
  size_t base = (size_t)row * HDIM + lane * 8;
  F4 xa = ((const F4*)(xc + base))[0], xb = ((const F4*)(xc + base))[1];
  F4 ma = ((const F4*)(m + base))[0],  mb_ = ((const F4*)(m + base))[1];
  float y[8], sm = 0.f, sq = 0.f;
#pragma unroll
  for (int j = 0; j < 8; ++j) {
    float xv = (j < 4) ? xa.v[j] : xb.v[j - 4];
    float mv = (j < 4) ? ma.v[j] : mb_.v[j - 4];
    float sg = (j < 4) ? r0 : r1;          // repeat(raw,4): c = h>>2
    float v = xv * sg + mv * (1.0f - sg);
    y[j] = v; sm += v; sq += v * v;
  }
  sm = wsum(sm); sq = wsum(sq);
  float mean = sm * (1.0f / HDIM);
  float rstd = rsqrtf(sq * (1.0f / HDIM) - mean * mean + 1e-5f);
  Pack8 pk; F4 o0, o1;
#pragma unroll
  for (int j = 0; j < 8; ++j) {
    int col = lane * 8 + j;
    float v = (y[j] - mean) * rstd * lns[col] + lnb[col];
    if (j < 4) o0.v[j] = v; else o1.v[j - 4] = v;
    pk.h[j] = f2bf(v);
  }
  ((F4*)(xc + base))[0] = o0; ((F4*)(xc + base))[1] = o1;
  *(U128*)(xcb + base) = pk.q;
}

// ---------------- readout ----------------
__global__ void pool_kernel(const float* __restrict__ xc, const int* __restrict__ batch,
                            float* __restrict__ pooled, float* __restrict__ cntg, int N) {
  int tid = blockIdx.x * blockDim.x + threadIdx.x;
  int node = tid >> 5, lane = tid & 31;
  if (node >= N) return;
  int g = batch[node];
  const float* xp = xc + (size_t)node * HDIM + lane * 8;
  float* pp = pooled + (size_t)g * HDIM + lane * 8;
#pragma unroll
  for (int j = 0; j < 8; ++j) atomicAdd(pp + j, xp[j]);
  if (lane == 0) atomicAdd(cntg + g, 1.0f);
}

__global__ void graph_out(const float* __restrict__ pooled, const float* __restrict__ cntg,
                          const float* __restrict__ Wout, const float* __restrict__ bout,
                          float* __restrict__ out) {
  int g = blockIdx.x, c = threadIdx.x;
  float inv = 1.0f / fmaxf(cntg[g], 1.0f);
  float acc = bout[c];
  for (int k = 0; k < HDIM; ++k) acc += pooled[(size_t)g * HDIM + k] * inv * Wout[(size_t)k * OUTDIM + c];
  out[(size_t)g * OUTDIM + c] = acc;
}

// ---------------- orchestration ----------------
extern "C" void kernel_launch(void* const* d_in, const int* in_sizes, int n_in,
                              void* d_out, int out_size, void* d_ws, size_t ws_size,
                              hipStream_t stream) {
  const float* x     = (const float*)d_in[0];
  const float* W_in  = (const float*)d_in[1];
  const float* b_in  = (const float*)d_in[2];
  const float* lis   = (const float*)d_in[3];
  const float* lib   = (const float*)d_in[4];
  const float* W_tm  = (const float*)d_in[5];
  const float* b_tm  = (const float*)d_in[6];
  const float* ln_s  = (const float*)d_in[7];
  const float* ln_b  = (const float*)d_in[8];
  const float* W_out = (const float*)d_in[9];
  const float* b_out = (const float*)d_in[10];
  const int*   ei    = (const int*)d_in[11];
  const int*   batch = (const int*)d_in[12];
  const int N = in_sizes[0] / INDIM;
  const int E = in_sizes[11] / 2;
  float* outF = (float*)d_out;

  char* ws = (char*)d_ws;
  size_t off = 0;
  auto alloc = [&](size_t bytes) { void* p = ws + off; off += (bytes + 15) & ~(size_t)15; return p; };
  float* xc     = (float*)alloc((size_t)N * HDIM * 4);
  float* m      = (float*)alloc((size_t)N * HDIM * 4);
  float* tm     = (float*)alloc((size_t)N * CDIM * 4);
  float* lg     = (float*)alloc((size_t)N * CDIM * 4);
  float* deg    = (float*)alloc((size_t)N * 4);
  float* pooled = (float*)alloc((size_t)NGRAPH * HDIM * 4);
  float* cntg   = (float*)alloc((size_t)NGRAPH * 4);
  u16* xcb = (u16*)alloc((size_t)N * HDIM * 2);
  u16* mb  = (u16*)alloc((size_t)N * HDIM * 2);
  u16* xib = (u16*)alloc((size_t)N * INDIM * 2);
  u16* wpi = (u16*)alloc((size_t)INDIM * HDIM * 2);
  u16* wpt = (u16*)alloc((size_t)4 * 2 * HDIM * CDIM * 2);
  u16* wpo = (u16*)alloc((size_t)HDIM * OUTDIM * 2);

  auto nb = [](size_t work, int bs) { return (unsigned)((work + bs - 1) / (size_t)bs); };

  zero_f<<<nb(N, 256), 256, 0, stream>>>(deg, N);
  zero_f<<<nb((size_t)N * CDIM, 256), 256, 0, stream>>>(tm, (size_t)N * CDIM);
  zero_f<<<nb((size_t)NGRAPH * HDIM, 256), 256, 0, stream>>>(pooled, (size_t)NGRAPH * HDIM);
  zero_f<<<nb(NGRAPH, 64), 64, 0, stream>>>(cntg, NGRAPH);

  pack_b<<<nb((INDIM / 32) * (HDIM / 16) * 32, 256), 256, 0, stream>>>(W_in, wpi, INDIM, HDIM);
  for (int l = 0; l < 4; ++l)
    pack_b<<<nb((512 / 32) * (CDIM / 16) * 32, 256), 256, 0, stream>>>(
        W_tm + (size_t)l * 512 * CDIM, wpt + (size_t)l * 512 * CDIM, 512, CDIM);
  pack_b<<<nb((HDIM / 32) * (OUTDIM / 16) * 32, 256), 256, 0, stream>>>(W_out, wpo, HDIM, OUTDIM);

  cvt_bf16<<<nb((size_t)N * INDIM, 256), 256, 0, stream>>>(x, xib, (size_t)N * INDIM);
  deg_kernel<<<nb(E, 256), 256, 0, stream>>>(ei, deg, E);

  // input stack: relu(x@W_in+b_in) -> m (tmp), then LN -> xc
  {
    size_t waves = (size_t)((N + 15) / 16) * (HDIM / 16);
    gemm_bf16_wmma<<<nb(waves * 32, 256), 256, 0, stream>>>(
        xib, xib, INDIM, INDIM, INDIM, wpi, b_in, m, HDIM, N, HDIM, INDIM);
  }
  ln_input<<<nb((size_t)N * 32, 256), 256, 0, stream>>>(m, lis, lib, xc, xcb, N);

  for (int l = 0; l < 4; ++l) {
    zero_f<<<nb((size_t)N * HDIM, 256), 256, 0, stream>>>(m, (size_t)N * HDIM);
    agg_scatter<<<nb((size_t)E * 32, 256), 256, 0, stream>>>(ei, xc, m, E);
    agg_finalize<<<nb((size_t)N * HDIM, 256), 256, 0, stream>>>(m, mb, xc, deg, (size_t)N * HDIM);
    {
      size_t waves = (size_t)((N + 15) / 16) * (CDIM / 16);
      gemm_bf16_wmma<<<nb(waves * 32, 256), 256, 0, stream>>>(
          xcb, mb, HDIM, HDIM, HDIM, wpt + (size_t)l * 512 * CDIM, b_tm + l * CDIM,
          lg, CDIM, N, CDIM, 2 * HDIM);
    }
    layer_post<<<nb((size_t)N * 32, 256), 256, 0, stream>>>(
        lg, tm, xc, xcb, m, ln_s + l * HDIM, ln_b + l * HDIM, N);
  }

  // lineout = xc @ W_out + b_out  (after the 64*64 xg block in d_out)
  {
    size_t waves = (size_t)((N + 15) / 16) * (OUTDIM / 16);
    gemm_bf16_wmma<<<nb(waves * 32, 256), 256, 0, stream>>>(
        xcb, xcb, HDIM, HDIM, HDIM, wpo, b_out, outF + NGRAPH * OUTDIM, OUTDIM, N, OUTDIM, HDIM);
  }
  pool_kernel<<<nb((size_t)N * 32, 256), 256, 0, stream>>>(xc, batch, pooled, cntg, N);
  graph_out<<<NGRAPH, OUTDIM, 0, stream>>>(pooled, cntg, W_out, b_out, outF);
}
